// AtomPairEmbedding_65335042506828
// MI455X (gfx1250) — matrix operand order repaired and checked by reading, not compiled, not run on hardware
//
#include <hip/hip_runtime.h>
#include <math.h>

#define NRES  8192
#define KTOT  48
#define KAA   32
#define PAIRC 128
#define HID   256
#define LNEPS 1e-5f

typedef __attribute__((ext_vector_type(16))) __bf16        v16bf;
typedef __attribute__((ext_vector_type(8)))  float         v8f;
typedef __attribute__((ext_vector_type(4)))  unsigned int  u32x4;
typedef __attribute__((ext_vector_type(2)))  unsigned int  u32x2;

union BF16Frag { v16bf v; u32x4 q[2]; unsigned short u[16]; };

__device__ __forceinline__ unsigned short f2bf(float f) {
    unsigned u = __float_as_uint(f);
    unsigned r = u + 0x7FFFu + ((u >> 16) & 1u);   // round-to-nearest-even
    return (unsigned short)(r >> 16);
}

// branch-free tanh via v_exp_f32 (no libm divergence)
__device__ __forceinline__ float gelu_tanh(float x) {
    float inner = 0.7978845608028654f * (x + 0.044715f * x * x * x);
    float ax = fabsf(inner);
    float e  = __expf(-2.0f * ax);
    float t  = (1.0f - e) / (1.0f + e);
    t = copysignf(t, inner);
    return 0.5f * x * (1.0f + t);
}

// ---------------------------------------------------------------- prep
__global__ void k_prep(const float* __restrict__ positions,
                       const float* __restrict__ all_atom_mask,
                       float* __restrict__ pos, float* __restrict__ maskf,
                       float* __restrict__ out_mask) {
    int t = blockIdx.x * blockDim.x + threadIdx.x;
    if (t >= NRES) return;
    const float* p = positions + ((long long)t * 14 + 1) * 3;
    pos[t * 3 + 0] = p[0];
    pos[t * 3 + 1] = p[1];
    pos[t * 3 + 2] = p[2];
    float m = (all_atom_mask[t * 14 + 1] > 0.f) ? 1.f : 0.f;
    maskf[t] = m;
    out_mask[t] = m;
}

// ---------------- weights f32 -> bf16, swizzled to fragment-ready tiles
// Tile (kk,nt) of a KxN weight: 512 bf16 laid out as [lane 0..31][half 0..15],
// element(lane l, half h) = W[(kk*32 + (l<16?0:8) + (h<8?h:h+8)) * N + nt*16 + (l&15)].
// A lane then loads its whole B fragment with two 16B loads.
__global__ void k_wswz(const float* __restrict__ W1, const float* __restrict__ W2,
                       unsigned short* __restrict__ W1s,
                       unsigned short* __restrict__ W2s) {
    int idx = blockIdx.x * blockDim.x + threadIdx.x;
    if (idx < PAIRC * HID) {
        int tile = idx >> 9, w = idx & 511;
        int l = w >> 4, h = w & 15;
        int kk = tile >> 4, nt = tile & 15;
        int krow = kk * 32 + ((l < 16) ? 0 : 8) + ((h < 8) ? h : h + 8);
        int col  = nt * 16 + (l & 15);
        W1s[idx] = f2bf(W1[krow * HID + col]);
    }
    if (idx < HID * HID) {
        int tile = idx >> 9, w = idx & 511;
        int l = w >> 4, h = w & 15;
        int kk = tile >> 4, nt = tile & 15;
        int krow = kk * 32 + ((l < 16) ? 0 : 8) + ((h < 8) ? h : h + 8);
        int col  = nt * 16 + (l & 15);
        W2s[idx] = f2bf(W2[krow * HID + col]);
    }
}

// ---------------------------------------------------------------- kNN
__global__ __launch_bounds__(256) void k_knn(
        const float* __restrict__ pos, const float* __restrict__ maskf,
        const int* __restrict__ is_aa,
        int* __restrict__ nb, float* __restrict__ out_nb) {
    __shared__ float         sdist[NRES];   // 32 KB
    __shared__ unsigned char sisaa[NRES];   // 8 KB
    __shared__ float         rwv[8];
    __shared__ int           riw[8];

    const int   i    = blockIdx.x;
    const int   tid  = threadIdx.x;
    const int   wave = tid >> 5;
    const int   lane = tid & 31;
    const float inf  = __builtin_inff();

    const float px = pos[i * 3 + 0], py = pos[i * 3 + 1], pz = pos[i * 3 + 2];
    const float mi = maskf[i];

    for (int j = tid; j < NRES; j += 256) {
        float dx = px - pos[j * 3 + 0];
        float dy = py - pos[j * 3 + 1];
        float dz = pz - pos[j * 3 + 2];
        float d  = sqrtf(fmaxf(dx * dx + dy * dy + dz * dz, 1e-12f));
        bool ok  = (mi > 0.f) && (maskf[j] > 0.f);
        sdist[j] = ok ? d : inf;
        sisaa[j] = (unsigned char)(is_aa[j] != 0);
    }
    __syncthreads();

    for (int k = 0; k < KTOT; ++k) {
        const bool wantAA = (k < KAA);
        float bv = inf; int bi = 0x7FFFFFFF;
        for (int j = tid; j < NRES; j += 256) {
            if ((sisaa[j] != 0) == wantAA) {
                float v = sdist[j];
                if (v < bv || (v == bv && j < bi)) { bv = v; bi = j; }
            }
        }
        // wave32 argmin reduction (no barriers)
        #pragma unroll
        for (int o = 16; o > 0; o >>= 1) {
            float ov = __shfl_down(bv, o, 32);
            int   oi = __shfl_down(bi, o, 32);
            if (ov < bv || (ov == bv && oi < bi)) { bv = ov; bi = oi; }
        }
        if (lane == 0) { rwv[wave] = bv; riw[wave] = bi; }
        __syncthreads();
        if (tid == 0) {
            float fv = rwv[0]; int fi = riw[0];
            #pragma unroll
            for (int w = 1; w < 8; ++w) {
                float ov = rwv[w]; int oi = riw[w];
                if (ov < fv || (ov == fv && oi < fi)) { fv = ov; fi = oi; }
            }
            int win = (fv < inf) ? fi : -1;
            nb[i * KTOT + k]     = win;
            out_nb[i * KTOT + k] = (float)win;
            if (win >= 0) sdist[win] = inf;
        }
        __syncthreads();
    }
}

// ------------------------------------------- edge features + pair = LN
__global__ __launch_bounds__(128) void k_edge(
        const float* __restrict__ pos,
        const int* __restrict__ is_aa,
        const int* __restrict__ chain_index,
        const int* __restrict__ residue_index,
        const int* __restrict__ nb,
        const float* __restrict__ Wp,
        const float* __restrict__ g, const float* __restrict__ b,
        float* __restrict__ out_pair) {
    __shared__ float red[128];
    __shared__ float sstat[2];
    const int e  = blockIdx.x;
    const int ch = threadIdx.x;
    const int i  = e / KTOT;
    int j  = nb[e];
    int jj = (j < 0) ? (NRES - 1) : j;   // JAX negative-index wrap

    float dx = pos[i * 3 + 0] - pos[jj * 3 + 0];
    float dy = pos[i * 3 + 1] - pos[jj * 3 + 1];
    float dz = pos[i * 3 + 2] - pos[jj * 3 + 2];
    float d  = sqrtf(fmaxf(dx * dx + dy * dy + dz * dz, 1e-12f));

    float typef = (is_aa[jj] != 0) ? 1.f : 0.f;
    bool  same_chain = (chain_index[i] == chain_index[jj]);
    float other      = same_chain ? 0.f : 1.f;
    float same_res   = (same_chain && (residue_index[i] == residue_index[jj])) ? 1.f : 0.f;

    float acc = 0.f;
    #pragma unroll
    for (int r = 0; r < 16; ++r) {
        float c = 2.0f + (20.0f / 15.0f) * (float)r;
        float t = (d - c) * (1.0f / 1.25f);
        acc = fmaf(__expf(-t * t), Wp[r * PAIRC + ch], acc);
    }
    acc = fmaf(typef,    Wp[16 * PAIRC + ch], acc);
    acc = fmaf(same_res, Wp[17 * PAIRC + ch], acc);
    acc = fmaf(other,    Wp[18 * PAIRC + ch], acc);

    // LayerNorm over 128 channels
    red[ch] = acc; __syncthreads();
    for (int s = 64; s > 0; s >>= 1) { if (ch < s) red[ch] += red[ch + s]; __syncthreads(); }
    if (ch == 0) sstat[0] = red[0] * (1.0f / 128.0f);
    __syncthreads();
    float diff = acc - sstat[0];
    red[ch] = diff * diff; __syncthreads();
    for (int s = 64; s > 0; s >>= 1) { if (ch < s) red[ch] += red[ch + s]; __syncthreads(); }
    if (ch == 0) sstat[1] = rsqrtf(red[0] * (1.0f / 128.0f) + LNEPS);
    __syncthreads();
    out_pair[(long long)e * PAIRC + ch] = diff * sstat[1] * g[ch] + b[ch];
}

// ------------------------------- per-residue MLP via WMMA + masked sum
__global__ __launch_bounds__(256) void k_mlp(
        const float* __restrict__ pair,
        const int* __restrict__ nb,
        const unsigned short* __restrict__ W1s,
        const unsigned short* __restrict__ W2s,
        const float* __restrict__ b1,
        const float* __restrict__ b2,
        float* __restrict__ pwOut) {
    __shared__ unsigned short Pbf[KTOT * PAIRC];  // 12 KB  pair tile (bf16)
    __shared__ unsigned short Hbf[KTOT * HID];    // 24 KB  hidden tile (bf16)
    __shared__ float          pwW[8 * HID];       // 8 KB   per-wave partial sums
    __shared__ float          mrow[KTOT];
    __shared__ float          s_nv;

    const int i    = blockIdx.x;
    const int tid  = threadIdx.x;
    const int wave = tid >> 5;
    const int lane = tid & 31;

    // stage pair tile: float4 loads, packed b64 LDS stores
    {
        const float4* p4 = (const float4*)(pair + (long long)i * KTOT * PAIRC);
        unsigned int* Pb32 = (unsigned int*)Pbf;
        for (int idx = tid; idx < KTOT * PAIRC / 4; idx += 256) {
            float4 f = p4[idx];
            u32x2 packed;
            packed.x = (unsigned)f2bf(f.x) | ((unsigned)f2bf(f.y) << 16);
            packed.y = (unsigned)f2bf(f.z) | ((unsigned)f2bf(f.w) << 16);
            *(u32x2*)&Pb32[idx * 2] = packed;
        }
    }
    if (tid < KTOT) mrow[tid] = (nb[i * KTOT + tid] != -1) ? 1.f : 0.f;
    for (int idx = tid; idx < 8 * HID; idx += 256) pwW[idx] = 0.f;
    __syncthreads();
    if (tid == 0) { float c = 0.f; for (int k = 0; k < KTOT; ++k) c += mrow[k]; s_nv = c; }

    const int halfsel = (lane < 16) ? 0 : 1;
    const int lrow    = lane & 15;

    // ---- stage 1: H = gelu(P(48x128) @ W1(128x256) + b1) ----
    for (int t = wave * 6; t < wave * 6 + 6; ++t) {
        const int mt = t >> 4, nt = t & 15;
        v8f acc = {};
        for (int kk = 0; kk < 4; ++kk) {
            BF16Frag A, B;
            const int arow = mt * 16 + lrow;
            const int ka   = kk * 32 + (halfsel ? 8 : 0);
            const unsigned short* ap = &Pbf[arow * PAIRC + ka];
            A.q[0] = *(const u32x4*)ap;            // halves: K = ka .. ka+7
            A.q[1] = *(const u32x4*)(ap + 16);     // halves: K = ka+16 .. ka+23
            const unsigned short* bt = W1s + ((((kk << 4) | nt) << 9) + (lane << 4));
            B.q[0] = *(const u32x4*)bt;
            B.q[1] = *(const u32x4*)(bt + 8);
            if (kk < 3) __builtin_prefetch(bt + (16 << 9), 0, 3);  // next k-tile
            acc = __builtin_amdgcn_wmma_f32_16x16x32_bf16(false, A.v, false, B.v,
                                                          (short)0, acc, false, false);
        }
        const int   col  = nt * 16 + lrow;
        const float bias = b1[col];
        const int   mb   = mt * 16 + (halfsel ? 8 : 0);
        #pragma unroll
        for (int r = 0; r < 8; ++r) {
            float x = acc[r] + bias;
            Hbf[(mb + r) * HID + col] = f2bf(gelu_tanh(x));
        }
    }
    __syncthreads();

    // ---- stage 2: MLP = H(48x256) @ W2(256x256); masked row-sum ----
    for (int t = wave * 6; t < wave * 6 + 6; ++t) {
        const int mt = t >> 4, nt = t & 15;
        v8f acc = {};
        for (int kk = 0; kk < 8; ++kk) {
            BF16Frag A, B;
            const int arow = mt * 16 + lrow;
            const int ka   = kk * 32 + (halfsel ? 8 : 0);
            const unsigned short* ap = &Hbf[arow * HID + ka];
            A.q[0] = *(const u32x4*)ap;
            A.q[1] = *(const u32x4*)(ap + 16);
            const unsigned short* bt = W2s + ((((kk << 4) | nt) << 9) + (lane << 4));
            B.q[0] = *(const u32x4*)bt;
            B.q[1] = *(const u32x4*)(bt + 8);
            if (kk < 7) __builtin_prefetch(bt + (16 << 9), 0, 3);
            acc = __builtin_amdgcn_wmma_f32_16x16x32_bf16(false, A.v, false, B.v,
                                                          (short)0, acc, false, false);
        }
        const int col = nt * 16 + lrow;
        const int mb  = mt * 16 + (halfsel ? 8 : 0);
        float s = 0.f;
        #pragma unroll
        for (int r = 0; r < 8; ++r) s += acc[r] * mrow[mb + r];
        s += __shfl_xor(s, 16, 32);          // combine the two row-halves of the tile
        if (lane < 16) pwW[wave * HID + col] += s;   // unique (wave,col) writer
    }
    __syncthreads();

    float tot = 0.f;
    #pragma unroll
    for (int w = 0; w < 8; ++w) tot += pwW[w * HID + tid];
    pwOut[(long long)i * HID + tid] = tot + b2[tid] * s_nv;  // + b2 * n_valid
}

// -------------------------------------------------- local = LN(... @ Wl)
__global__ __launch_bounds__(128) void k_local(
        const float* __restrict__ pw,
        const int* __restrict__ is_aa,
        const int* __restrict__ aa,
        const float* __restrict__ Wl,
        const float* __restrict__ g, const float* __restrict__ b,
        float* __restrict__ out_local) {
    __shared__ float sin_[HID];
    __shared__ float red[128];
    __shared__ float sstat[2];
    const int i  = blockIdx.x;
    const int ch = threadIdx.x;
    sin_[ch]       = pw[(long long)i * HID + ch];
    sin_[ch + 128] = pw[(long long)i * HID + ch + 128];
    __syncthreads();
    float acc = 0.f;
    for (int k = 0; k < HID; ++k)
        acc = fmaf(sin_[k], Wl[k * PAIRC + ch], acc);
    float isf = (is_aa[i] != 0) ? 1.f : 0.f;
    acc = fmaf(isf, Wl[256 * PAIRC + ch], acc);
    int a = aa[i];
    acc += Wl[(257 + a) * PAIRC + ch];
    // LayerNorm over 128
    red[ch] = acc; __syncthreads();
    for (int s = 64; s > 0; s >>= 1) { if (ch < s) red[ch] += red[ch + s]; __syncthreads(); }
    if (ch == 0) sstat[0] = red[0] * (1.f / 128.f);
    __syncthreads();
    float diff = acc - sstat[0];
    red[ch] = diff * diff; __syncthreads();
    for (int s = 64; s > 0; s >>= 1) { if (ch < s) red[ch] += red[ch + s]; __syncthreads(); }
    if (ch == 0) sstat[1] = rsqrtf(red[0] * (1.f / 128.f) + LNEPS);
    __syncthreads();
    out_local[(long long)i * PAIRC + ch] = diff * sstat[1] * g[ch] + b[ch];
}

// ---------------------------------------------------------------- host
extern "C" void kernel_launch(void* const* d_in, const int* in_sizes, int n_in,
                              void* d_out, int out_size, void* d_ws, size_t ws_size,
                              hipStream_t stream) {
    (void)in_sizes; (void)n_in; (void)out_size; (void)ws_size;

    const float* positions      = (const float*)d_in[0];
    const float* all_atom_mask  = (const float*)d_in[1];
    const int*   aa             = (const int*)d_in[2];
    const int*   chain_index    = (const int*)d_in[3];
    const int*   residue_index  = (const int*)d_in[4];
    const int*   is_aa          = (const int*)d_in[5];
    const float* Wp             = (const float*)d_in[6];
    const float* ln1_g          = (const float*)d_in[7];
    const float* ln1_b          = (const float*)d_in[8];
    const float* W1             = (const float*)d_in[9];
    const float* b1             = (const float*)d_in[10];
    const float* W2             = (const float*)d_in[11];
    const float* b2             = (const float*)d_in[12];
    const float* Wl             = (const float*)d_in[13];
    const float* ln2_g          = (const float*)d_in[14];
    const float* ln2_b          = (const float*)d_in[15];

    float* out       = (float*)d_out;
    float* out_local = out;                                         // N*128
    float* out_pair  = out + (size_t)NRES * PAIRC;                  // N*48*128
    float* out_nb    = out_pair + (size_t)NRES * KTOT * PAIRC;      // N*48
    float* out_mask  = out_nb + (size_t)NRES * KTOT;                // N

    char*  ws  = (char*)d_ws;
    size_t off = 0;
    auto walloc = [&](size_t bytes) -> void* {
        void* p = ws + off;
        off += bytes;
        off = (off + 255) & ~(size_t)255;
        return p;
    };
    float*          pos   = (float*)walloc((size_t)NRES * 3 * sizeof(float));
    float*          maskf = (float*)walloc((size_t)NRES * sizeof(float));
    int*            nb    = (int*)  walloc((size_t)NRES * KTOT * sizeof(int));
    float*          pw    = (float*)walloc((size_t)NRES * HID * sizeof(float));
    unsigned short* W1s   = (unsigned short*)walloc((size_t)PAIRC * HID * 2);
    unsigned short* W2s   = (unsigned short*)walloc((size_t)HID * HID * 2);

    k_prep <<<(NRES + 255) / 256, 256, 0, stream>>>(positions, all_atom_mask, pos, maskf, out_mask);
    k_wswz <<<(HID * HID + 255) / 256, 256, 0, stream>>>(W1, W2, W1s, W2s);
    k_knn  <<<NRES, 256, 0, stream>>>(pos, maskf, is_aa, nb, out_nb);
    k_edge <<<NRES * KTOT, 128, 0, stream>>>(pos, is_aa, chain_index, residue_index,
                                             nb, Wp, ln1_g, ln1_b, out_pair);
    k_mlp  <<<NRES, 256, 0, stream>>>(out_pair, nb, W1s, W2s, b1, b2, pw);
    k_local<<<NRES, 128, 0, stream>>>(pw, is_aa, aa, Wl, ln2_g, ln2_b, out_local);
}
